// GNNAUX_54795192762424
// MI455X (gfx1250) — matrix operand order repaired and checked by reading, not compiled, not run on hardware
//
#include <hip/hip_runtime.h>
#include <hip/hip_bf16.h>

// ---------------------------------------------------------------------------
// GNN + kernel-aggregate for MI455X (gfx1250, wave32, WMMA)
// Dominant op: z = (kappa @ h2) / rowsum(kappa), kappa = exp(-(si-sj)^2),
// N = 16384, D = 64  ->  34.4 GFLOP GEMM + 268M exp, all L2-resident.
// GEMM runs on v_wmma_f32_16x16x32_bf16 (bf16 in, fp32 accumulate).
// h2 kept pre-transposed in global (h2bfT[d][j]); LDS staging is a pure
// row copy, software-pipelined (register prefetch) and double-buffered with
// one barrier per 32-wide j tile. j-range split 4-way (blockIdx.y) for wave
// parallelism; partial numerators/row-sums combined in the classifier.
// ---------------------------------------------------------------------------

typedef __bf16 v16bf __attribute__((ext_vector_type(16)));
typedef float  v8f   __attribute__((ext_vector_type(8)));

#define N_NODES 16384
#define N_EDGES 262144
#define F_IN    128
#define DIM     64
#define N_CLS   40
#define EPS_F   1e-8f

// ---------------- utility: zero a float range (grid-stride) ----------------
__global__ void k_zero(float* __restrict__ p, int n) {
    for (int i = blockIdx.x * blockDim.x + threadIdx.x; i < n;
         i += gridDim.x * blockDim.x)
        p[i] = 0.0f;
}

// ---------------- scatter-add of x[src] rows into agg[dst] (128 feats) -----
__global__ void k_scatter1(const int* __restrict__ ei, const float* __restrict__ x,
                           float* __restrict__ agg, float* __restrict__ cnt) {
    int t = blockIdx.x * blockDim.x + threadIdx.x;
    if (t >= N_EDGES * 32) return;
    int e  = t >> 5;
    int f4 = (t & 31) << 2;                 // 4 features per thread, 32 thr/edge
    int s  = ei[e];
    int d  = ei[N_EDGES + e];
    const float4 v = *(const float4*)&x[s * F_IN + f4];
    float* a = &agg[d * F_IN + f4];
    atomicAdd(a + 0, v.x); atomicAdd(a + 1, v.y);
    atomicAdd(a + 2, v.z); atomicAdd(a + 3, v.w);
    if (f4 == 0) atomicAdd(&cnt[d], 1.0f);
}

// ---------------- SAGE layer 1: h1 = relu(mean_agg@W1l + b1 + x@W1r) -------
__global__ void k_sage1(const float* __restrict__ x, const float* __restrict__ agg,
                        const float* __restrict__ cnt, const float* __restrict__ W1l,
                        const float* __restrict__ b1, const float* __restrict__ W1r,
                        float* __restrict__ h1) {
    int i = blockIdx.x, d = threadIdx.x;        // 64 threads/node
    float inv = 1.0f / fmaxf(cnt[i], 1.0f);
    float acc = b1[d];
    const float* ar = &agg[i * F_IN];
    const float* xr = &x[i * F_IN];
#pragma unroll 8
    for (int k = 0; k < F_IN; ++k)
        acc += ar[k] * inv * W1l[k * DIM + d] + xr[k] * W1r[k * DIM + d];
    h1[i * DIM + d] = fmaxf(acc, 0.0f);
}

// ---------------- scatter-add of h1[src] rows into agg2[dst] (64 feats) ----
__global__ void k_scatter2(const int* __restrict__ ei, const float* __restrict__ h1,
                           float* __restrict__ agg2) {
    int t = blockIdx.x * blockDim.x + threadIdx.x;
    if (t >= N_EDGES * 16) return;
    int e  = t >> 4;
    int f4 = (t & 15) << 2;
    int s  = ei[e];
    int d  = ei[N_EDGES + e];
    const float4 v = *(const float4*)&h1[s * DIM + f4];
    float* a = &agg2[d * DIM + f4];
    atomicAdd(a + 0, v.x); atomicAdd(a + 1, v.y);
    atomicAdd(a + 2, v.z); atomicAdd(a + 3, v.w);
}

// ---- SAGE layer 2 (no relu); emit fp32 h2 and TRANSPOSED bf16 h2bfT -------
__global__ void k_sage2(const float* __restrict__ h1, const float* __restrict__ agg,
                        const float* __restrict__ cnt, const float* __restrict__ W2l,
                        const float* __restrict__ b2, const float* __restrict__ W2r,
                        float* __restrict__ h2, unsigned short* __restrict__ h2bfT) {
    int i = blockIdx.x, d = threadIdx.x;        // 64 threads/node
    float inv = 1.0f / fmaxf(cnt[i], 1.0f);
    float acc = b2[d];
    const float* ar = &agg[i * DIM];
    const float* hr = &h1[i * DIM];
#pragma unroll 8
    for (int k = 0; k < DIM; ++k)
        acc += ar[k] * inv * W2l[k * DIM + d] + hr[k] * W2r[k * DIM + d];
    h2[i * DIM + d] = acc;
    union { __bf16 b; unsigned short u; } cv;
    cv.b = (__bf16)acc;
    h2bfT[d * N_NODES + i] = cv.u;              // transposed: [DIM][N]
}

// ---------------- cosine-similarity score vs normalized aux ----------------
__global__ void k_score(const float* __restrict__ h2, const float* __restrict__ aux,
                        float* __restrict__ score) {
    int i = blockIdx.x * blockDim.x + threadIdx.x;
    if (i >= N_NODES) return;
    float an = 0.0f, hn = 0.0f, dt = 0.0f;
#pragma unroll 8
    for (int k = 0; k < DIM; ++k) {
        float a = aux[k], h = h2[i * DIM + k];
        an += a * a; hn += h * h; dt += a * h;
    }
    score[i] = dt / (fmaxf(sqrtf(an), EPS_F) * fmaxf(sqrtf(hn), EPS_F));
}

// ---------------------------------------------------------------------------
// Kernel aggregate (partial): for j-slice s = blockIdx.y,
//   zpart[s]  += kappa_tile @ h2_tile   (unnormalized)
//   rspart[s] += rowsum(kappa_tile)
// 128 threads = 4 waves; each wave owns a 16-row i-tile (64 rows/block).
// Register-prefetch software pipeline + double-buffered LDS, 1 barrier/tile.
// ---------------------------------------------------------------------------
#define BJ      32
#define TSTR    40   // padded LDS row stride (ushorts): 80B, 16B-aligned
#define BLOCK_T 128
#define JSPLIT  4
#define NJ_PER  (N_NODES / JSPLIT)     // 4096 j per block
#define JTILES  (NJ_PER / BJ)          // 128 tiles

__global__ void __launch_bounds__(BLOCK_T) k_kagg(const float* __restrict__ score,
                                                  const unsigned short* __restrict__ h2bfT,
                                                  float* __restrict__ zpart,
                                                  float* __restrict__ rspart) {
    __shared__ __align__(16) unsigned short h2Ts[2][DIM * TSTR];  // [buf][d][jl]
    __shared__ __align__(16) float sjs[2][BJ];

    const int tid  = threadIdx.x;
    const int wave = tid >> 5;
    const int lane = tid & 31;
    const int hi   = lane >> 4;       // lane half
    const int m    = lane & 15;       // row (A) / column (B,C) within tile
    const int row0 = blockIdx.x * 64 + wave * 16;
    const int jbase = blockIdx.y * NJ_PER;
    const float si = score[row0 + m];

    // staging slots (two 16B rows per thread per tile)
    const int d0  = tid >> 2,            jq0 = (tid & 3) * 8;
    const int d1  = (tid + BLOCK_T) >> 2, jq1 = ((tid + BLOCK_T) & 3) * 8;
    const unsigned short* g0 = &h2bfT[d0 * N_NODES + jbase + jq0];
    const unsigned short* g1 = &h2bfT[d1 * N_NODES + jbase + jq1];

    // prefetch tile 0 into registers
    uint4 r0 = *(const uint4*)g0;
    uint4 r1 = *(const uint4*)g1;
    float scp = (tid < BJ) ? score[jbase + tid] : 0.0f;

    v8f acc0 = {}, acc1 = {}, acc2 = {}, acc3 = {};
    float rs0 = 0.0f, rs1 = 0.0f, rs2 = 0.0f, rs3 = 0.0f;  // kappa row-sum parts

    int buf = 0;
    for (int t = 0; t < JTILES; ++t) {
        // ---- commit prefetched tile to LDS ----
        *(uint4*)&h2Ts[buf][d0 * TSTR + jq0] = r0;
        *(uint4*)&h2Ts[buf][d1 * TSTR + jq1] = r1;
        if (tid < BJ) sjs[buf][tid] = scp;
        __syncthreads();

        // ---- prefetch next tile (overlaps with compute below) ----
        if (t + 1 < JTILES) {
            const int joff = (t + 1) * BJ;
            r0 = *(const uint4*)(g0 + joff);
            r1 = *(const uint4*)(g1 + joff);
            if (tid < BJ) scp = score[jbase + joff + tid];
        }

        // ---- build A (16x32 bf16): element e has K = e + 16*hi
        const float4 s0 = *(const float4*)&sjs[buf][hi * 16];
        const float4 s1 = *(const float4*)&sjs[buf][hi * 16 + 4];
        const float4 s2 = *(const float4*)&sjs[buf][hi * 16 + 8];
        const float4 s3 = *(const float4*)&sjs[buf][hi * 16 + 12];
        const float sv[16] = { s0.x, s0.y, s0.z, s0.w,  s1.x, s1.y, s1.z, s1.w,
                               s2.x, s2.y, s2.z, s2.w,  s3.x, s3.y, s3.z, s3.w };
        v16bf a;
        float kp[16];
#pragma unroll
        for (int e = 0; e < 16; ++e) {
            float dd = sv[e] - si;
            kp[e] = __expf(-dd * dd);           // SIGMA = 1
            a[e] = (__bf16)kp[e];
        }
#pragma unroll
        for (int e = 0; e < 16; e += 4) {       // 4 independent add chains
            rs0 += kp[e + 0]; rs1 += kp[e + 1];
            rs2 += kp[e + 2]; rs3 += kp[e + 3];
        }

        // ---- B operands: contiguous 32B per lane from LDS tile ----
        const v16bf b0 = *(const v16bf*)&h2Ts[buf][(0 * 16 + m) * TSTR + hi * 16];
        const v16bf b1 = *(const v16bf*)&h2Ts[buf][(1 * 16 + m) * TSTR + hi * 16];
        const v16bf b2 = *(const v16bf*)&h2Ts[buf][(2 * 16 + m) * TSTR + hi * 16];
        const v16bf b3 = *(const v16bf*)&h2Ts[buf][(3 * 16 + m) * TSTR + hi * 16];

        acc0 = __builtin_amdgcn_wmma_f32_16x16x32_bf16(false, a, false, b0, (short)0, acc0, false, false);
        acc1 = __builtin_amdgcn_wmma_f32_16x16x32_bf16(false, a, false, b1, (short)0, acc1, false, false);
        acc2 = __builtin_amdgcn_wmma_f32_16x16x32_bf16(false, a, false, b2, (short)0, acc2, false, false);
        acc3 = __builtin_amdgcn_wmma_f32_16x16x32_bf16(false, a, false, b3, (short)0, acc3, false, false);

        buf ^= 1;
    }

    // partial rowsum: combine the two lane-halves holding complementary K slots
    float rs  = (rs0 + rs1) + (rs2 + rs3);
    float rsT = rs + __shfl_xor(rs, 16, 32);    // lane L: total for row (L&15)
    if (hi == 0)
        rspart[blockIdx.y * N_NODES + row0 + m] = rsT;

    // store unnormalized partial numerator
    float* zp = zpart + (size_t)blockIdx.y * N_NODES * DIM;
#pragma unroll
    for (int r = 0; r < 8; ++r) {
        int row = row0 + r + hi * 8;
        zp[row * DIM + 0 * 16 + m] = acc0[r];
        zp[row * DIM + 1 * 16 + m] = acc1[r];
        zp[row * DIM + 2 * 16 + m] = acc2[r];
        zp[row * DIM + 3 * 16 + m] = acc3[r];
    }
}

// ------- classifier: out = [h2, (sum zpart)/(sum rspart)] @ Wc + bc --------
__global__ void k_cls(const float* __restrict__ h2, const float* __restrict__ zp,
                      const float* __restrict__ rsp, const float* __restrict__ Wc,
                      const float* __restrict__ bc, float* __restrict__ out) {
    int t = blockIdx.x * blockDim.x + threadIdx.x;
    if (t >= N_NODES * N_CLS) return;
    int i = t / N_CLS, c = t % N_CLS;
    float rsum = 0.0f;
#pragma unroll
    for (int s = 0; s < JSPLIT; ++s) rsum += rsp[s * N_NODES + i];
    const float rinv = 1.0f / rsum;
    float acc = bc[c];
#pragma unroll 8
    for (int d = 0; d < DIM; ++d) {
        float zv = 0.0f;
#pragma unroll
        for (int s = 0; s < JSPLIT; ++s)
            zv += zp[(size_t)s * N_NODES * DIM + i * DIM + d];
        acc += h2[i * DIM + d] * Wc[d * N_CLS + c];
        acc += zv * rinv       * Wc[(DIM + d) * N_CLS + c];
    }
    out[t] = acc;
}

// ---------------------------------------------------------------------------
extern "C" void kernel_launch(void* const* d_in, const int* in_sizes, int n_in,
                              void* d_out, int out_size, void* d_ws, size_t ws_size,
                              hipStream_t stream) {
    const float* x   = (const float*)d_in[0];
    const int*   ei  = (const int*)  d_in[1];
    const float* W1l = (const float*)d_in[2];
    const float* b1  = (const float*)d_in[3];
    const float* W1r = (const float*)d_in[4];
    const float* W2l = (const float*)d_in[5];
    const float* b2  = (const float*)d_in[6];
    const float* W2r = (const float*)d_in[7];
    const float* aux = (const float*)d_in[8];
    const float* Wc  = (const float*)d_in[9];
    const float* bc  = (const float*)d_in[10];
    float* out = (float*)d_out;

    // workspace layout (cnt/agg1/agg2 contiguous so one zero pass covers them)
    float* ws   = (float*)d_ws;
    size_t o    = 0;
    float* cnt  = ws + o;  o += (size_t)N_NODES;
    float* agg1 = ws + o;  o += (size_t)N_NODES * F_IN;
    float* agg2 = ws + o;  o += (size_t)N_NODES * DIM;
    float* h1   = ws + o;  o += (size_t)N_NODES * DIM;
    float* h2   = ws + o;  o += (size_t)N_NODES * DIM;
    float* scr  = ws + o;  o += (size_t)N_NODES;
    unsigned short* h2bfT = (unsigned short*)(ws + o);   // [DIM][N] bf16
    o += (size_t)N_NODES * DIM / 2;                      // ushort = half float
    float* zpart  = ws + o;  o += (size_t)JSPLIT * N_NODES * DIM;
    float* rspart = ws + o;  o += (size_t)JSPLIT * N_NODES;

    const int zeroN = N_NODES * (1 + F_IN + DIM);
    k_zero<<<1024, 256, 0, stream>>>(cnt, zeroN);

    k_scatter1<<<(N_EDGES * 32) / 256, 256, 0, stream>>>(ei, x, agg1, cnt);
    k_sage1<<<N_NODES, DIM, 0, stream>>>(x, agg1, cnt, W1l, b1, W1r, h1);

    k_scatter2<<<(N_EDGES * 16) / 256, 256, 0, stream>>>(ei, h1, agg2);
    k_sage2<<<N_NODES, DIM, 0, stream>>>(h1, agg2, cnt, W2l, b2, W2r, h2, h2bfT);

    k_score<<<N_NODES / 256, 256, 0, stream>>>(h2, aux, scr);

    dim3 gk(N_NODES / 64, JSPLIT);
    k_kagg<<<gk, BLOCK_T, 0, stream>>>(scr, h2bfT, zpart, rspart);

    k_cls<<<(N_NODES * N_CLS + 255) / 256, 256, 0, stream>>>(h2, zpart, rspart, Wc, bc, out);
}
